// RNNStack_34291018891898
// MI455X (gfx1250) — compile-verified, matrix operand order checked by reading, not run on hardware
//
#include <hip/hip_runtime.h>
#include <hip/hip_bf16.h>

typedef __bf16 bf16_t;
typedef __attribute__((ext_vector_type(8)))  __bf16 v8bf;
typedef __attribute__((ext_vector_type(16))) __bf16 v16bf;
typedef __attribute__((ext_vector_type(8)))  float  v8f;

#define T_SEQ 8192
#define DIM   512      // D == H0 == H1
#define NC    2048     // 4*H
#define NWG   16       // persistent workgroups in recurrent scan
#define SCAN_THREADS 256

// ------------------------------------------------------------------
// init: zero h double-buffers and barrier counters (ws is poisoned)
// ------------------------------------------------------------------
__global__ void init_kernel(float* hbufA, float* hbufB, unsigned* counters) {
  int i = blockIdx.x * blockDim.x + threadIdx.x;
  if (i < 2 * DIM) { hbufA[i] = 0.f; hbufB[i] = 0.f; }
  if (i < 8) counters[i] = 0u;
}

// ------------------------------------------------------------------
// fp32 -> (bf16 hi, bf16 lo) split, elementwise
// ------------------------------------------------------------------
__global__ void split_kernel(const float* __restrict__ x,
                             bf16_t* __restrict__ hi, bf16_t* __restrict__ lo, int n) {
  int i = blockIdx.x * blockDim.x + threadIdx.x;
  if (i < n) {
    float v = x[i];
    bf16_t h = (bf16_t)v;
    hi[i] = h;
    lo[i] = (bf16_t)(v - (float)h);
  }
}

// W[DIM][NC] row-major  ->  Bt[NC][DIM] (transposed) hi/lo bf16
__global__ void splitT_kernel(const float* __restrict__ w,
                              bf16_t* __restrict__ hi, bf16_t* __restrict__ lo) {
  int i = blockIdx.x * blockDim.x + threadIdx.x;   // over NC*DIM outputs
  if (i < NC * DIM) {
    int n = i >> 9;          // output row (= column of W), DIM=512
    int k = i & (DIM - 1);
    float v = w[(size_t)k * NC + n];
    bf16_t h = (bf16_t)v;
    hi[i] = h;
    lo[i] = (bf16_t)(v - (float)h);
  }
}

// ------------------------------------------------------------------
// Z[T_SEQ][NC] = A[T_SEQ][DIM] @ B[DIM][NC] + bias, via bf16x3 WMMA.
// A given row-major bf16 (hi/lo); B given transposed [NC][DIM] bf16 (hi/lo)
// so both fragments are contiguous 16-byte loads per lane.
// One 16x16 C tile per wave, 8 waves per block.
// ------------------------------------------------------------------
__global__ __launch_bounds__(256) void gemm_kernel(
    const bf16_t* __restrict__ Ahi, const bf16_t* __restrict__ Alo,
    const bf16_t* __restrict__ Bhi, const bf16_t* __restrict__ Blo,
    const float* __restrict__ bias, float* __restrict__ Z) {
  const int wave = threadIdx.x >> 5;
  const int lane = threadIdx.x & 31;
  const int tile = blockIdx.x * 8 + wave;
  const int ntn  = NC / 16;                 // 128 tiles along N
  const int mt   = tile / ntn;
  const int nt   = tile - mt * ntn;
  const int r    = lane & 15;
  const int sel  = lane >> 4;               // half-wave selector

  // 16-bit A layout: lane<16 holds row r, K {k0..k0+7, k0+16..k0+23};
  // lane>=16 holds row r, K {k0+8..k0+15, k0+24..k0+31}. B mirrors with N<->M.
  const bf16_t* aH = Ahi + (size_t)(mt * 16 + r) * DIM;
  const bf16_t* aL = Alo + (size_t)(mt * 16 + r) * DIM;
  const bf16_t* bH = Bhi + (size_t)(nt * 16 + r) * DIM;
  const bf16_t* bL = Blo + (size_t)(nt * 16 + r) * DIM;

  v8f acc = {};
  for (int k0 = 0; k0 < DIM; k0 += 32) {
    const int o0 = k0 + sel * 8;
    const int o1 = k0 + 16 + sel * 8;
    v8bf ah0 = *(const v8bf*)(aH + o0), ah1 = *(const v8bf*)(aH + o1);
    v8bf al0 = *(const v8bf*)(aL + o0), al1 = *(const v8bf*)(aL + o1);
    v8bf bh0 = *(const v8bf*)(bH + o0), bh1 = *(const v8bf*)(bH + o1);
    v8bf bl0 = *(const v8bf*)(bL + o0), bl1 = *(const v8bf*)(bL + o1);
    v16bf AH, AL, BH, BL;
#pragma unroll
    for (int i = 0; i < 8; ++i) {
      AH[i] = ah0[i]; AH[i + 8] = ah1[i];
      AL[i] = al0[i]; AL[i + 8] = al1[i];
      BH[i] = bh0[i]; BH[i + 8] = bh1[i];
      BL[i] = bl0[i]; BL[i + 8] = bl1[i];
    }
    // bf16x3: AhBh + AlBh + AhBl (AlBl term negligible), fp32 accumulate
    acc = __builtin_amdgcn_wmma_f32_16x16x32_bf16(false, AH, false, BH, (short)0, acc, false, false);
    acc = __builtin_amdgcn_wmma_f32_16x16x32_bf16(false, AL, false, BH, (short)0, acc, false, false);
    acc = __builtin_amdgcn_wmma_f32_16x16x32_bf16(false, AH, false, BL, (short)0, acc, false, false);
  }

  // C/D layout: VGPR i -> row (mt*16 + sel*8 + i), col (nt*16 + r)
  const int col = nt * 16 + r;
  const int row0 = mt * 16 + sel * 8;
  const float b = bias[col];
#pragma unroll
  for (int i = 0; i < 8; ++i)
    Z[(size_t)(row0 + i) * NC + col] = acc[i] + b;
}

// ------------------------------------------------------------------
// Persistent recurrent scan. 16 WGs; WG k owns output indices
// j in [32k, 32k+32) across all 4 gates -> 128 Wh columns, its
// 512x128 fp32 slice lives in LDS (256 KB) for the whole sequence.
// One agent-scope barrier per step; h double-buffered in global.
// ------------------------------------------------------------------
__global__ __launch_bounds__(SCAN_THREADS) void scan_kernel(
    const float* __restrict__ Z,     // [T_SEQ][NC], includes x@Wi + bias
    const float* __restrict__ Wh,    // [DIM][NC] fp32
    float* __restrict__ hs_out,      // [T_SEQ][DIM] or nullptr
    float* hbuf,                     // [2][DIM] double buffer (pre-zeroed)
    unsigned* counter,               // monotonic barrier counter (pre-zeroed)
    const float* __restrict__ xs_for_n,
    float* __restrict__ d_out, int write_out) {
  __shared__ float whS[DIM * 128];   // 256 KB  (CDNA5: 320 KB LDS per WG)
  __shared__ float h_lds[DIM];
  __shared__ float zpart[8 * 128];
  __shared__ float zv[128];
  __shared__ float c_lds[32];

  const int tid = threadIdx.x;
  const int wg  = blockIdx.x;

  // Stage this WG's 128 Wh columns: local col c -> gate (c>>5), j = wg*32 + (c&31)
  for (int idx = tid; idx < DIM * 128; idx += SCAN_THREADS) {
    int rr = idx >> 7, cc = idx & 127;
    int col = ((cc >> 5) << 9) + wg * 32 + (cc & 31);
    whS[idx] = Wh[(size_t)rr * NC + col];
  }
  if (tid < 32) c_lds[tid] = 0.f;
  int N = 0;
  if (write_out) N = (int)xs_for_n[(size_t)(T_SEQ - 1) * DIM];
  __syncthreads();

  const int cg = tid & 31;   // column group: 4 consecutive local cols
  const int rc = tid >> 5;   // row chunk: 64 rows

  for (int t = 0; t < T_SEQ; ++t) {
    const int p = t & 1;
    // pull current h (written by all WGs last step, or zero-init)
    for (int i = tid; i < DIM; i += SCAN_THREADS)
      h_lds[i] = __hip_atomic_load(&hbuf[p * DIM + i], __ATOMIC_RELAXED,
                                   __HIP_MEMORY_SCOPE_AGENT);
    __syncthreads();

    // GEMV partials: thread does 64 rows x 4 cols from LDS
    float a0 = 0.f, a1 = 0.f, a2 = 0.f, a3 = 0.f;
    const int rbase = rc * 64;
#pragma unroll 4
    for (int r2 = 0; r2 < 64; ++r2) {
      const int rr = rbase + r2;
      const float hv = h_lds[rr];
      const float4 w = *(const float4*)&whS[rr * 128 + cg * 4];
      a0 += hv * w.x; a1 += hv * w.y; a2 += hv * w.z; a3 += hv * w.w;
    }
    *(float4*)&zpart[rc * 128 + cg * 4] = make_float4(a0, a1, a2, a3);
    __syncthreads();

    if (tid < 128) {
      float z = 0.f;
#pragma unroll
      for (int rr = 0; rr < 8; ++rr) z += zpart[rr * 128 + tid];
      const int col = ((tid >> 5) << 9) + wg * 32 + (tid & 31);
      z += Z[(size_t)t * NC + col];
      zv[tid] = z;
    }
    __syncthreads();

    if (tid < 32) {
      const float ig = 1.f / (1.f + expf(-zv[tid]));
      const float fg = 1.f / (1.f + expf(-zv[32 + tid]));
      const float gg = tanhf(zv[64 + tid]);
      const float og = 1.f / (1.f + expf(-zv[96 + tid]));
      const float cv = fg * c_lds[tid] + ig * gg;
      c_lds[tid] = cv;
      const float hv = og * tanhf(cv);
      const int J = wg * 32 + tid;
      __hip_atomic_store(&hbuf[(p ^ 1) * DIM + J], hv, __ATOMIC_RELAXED,
                         __HIP_MEMORY_SCOPE_AGENT);
      if (hs_out) hs_out[(size_t)t * DIM + J] = hv;
      if (write_out && t == N - 1) d_out[J] = hv;
    }
    __syncthreads();

    // inter-WG barrier (monotonic counter, release/acquire at device scope)
    if (tid == 0) {
      __hip_atomic_fetch_add(counter, 1u, __ATOMIC_RELEASE, __HIP_MEMORY_SCOPE_AGENT);
      const unsigned goal = (unsigned)(t + 1) * NWG;
      while (__hip_atomic_load(counter, __ATOMIC_ACQUIRE, __HIP_MEMORY_SCOPE_AGENT) < goal)
        __builtin_amdgcn_s_sleep(2);
    }
    __syncthreads();
  }
}

// ------------------------------------------------------------------
extern "C" void kernel_launch(void* const* d_in, const int* in_sizes, int n_in,
                              void* d_out, int out_size, void* d_ws, size_t ws_size,
                              hipStream_t stream) {
  const float* xs  = (const float*)d_in[0];
  const float* Wi0 = (const float*)d_in[1];
  const float* Wh0 = (const float*)d_in[2];
  const float* b0  = (const float*)d_in[3];
  const float* Wi1 = (const float*)d_in[4];
  const float* Wh1 = (const float*)d_in[5];
  const float* b1  = (const float*)d_in[6];
  float* out = (float*)d_out;

  // workspace layout (bytes)
  char* ws = (char*)d_ws;
  float*  Z     = (float*)(ws);                    // 8192*2048*4 = 64 MiB
  float*  hs0   = (float*)(ws + 67108864);         // 8192*512*4  = 16 MiB
  bf16_t* Ahi   = (bf16_t*)(ws + 83886080);        // 8 MiB
  bf16_t* Alo   = (bf16_t*)(ws + 92274688);        // 8 MiB
  bf16_t* Bhi   = (bf16_t*)(ws + 100663296);       // 2 MiB
  bf16_t* Blo   = (bf16_t*)(ws + 102760448);       // 2 MiB
  float*  hbufA = (float*)(ws + 104857600);        // 4 KiB
  float*  hbufB = (float*)(ws + 104861696);        // 4 KiB
  unsigned* ctr = (unsigned*)(ws + 104865792);

  const int nA = T_SEQ * DIM;
  const int gemmBlocks = (T_SEQ / 16) * (NC / 16) / 8;   // 8192

  init_kernel<<<4, 256, 0, stream>>>(hbufA, hbufB, ctr);

  // ---- layer 0 ----
  split_kernel<<<(nA + 255) / 256, 256, 0, stream>>>(xs, Ahi, Alo, nA);
  splitT_kernel<<<(NC * DIM + 255) / 256, 256, 0, stream>>>(Wi0, Bhi, Blo);
  gemm_kernel<<<gemmBlocks, 256, 0, stream>>>(Ahi, Alo, Bhi, Blo, b0, Z);
  scan_kernel<<<NWG, SCAN_THREADS, 0, stream>>>(Z, Wh0, hs0, hbufA, &ctr[0], xs, out, 0);

  // ---- layer 1 ----
  split_kernel<<<(nA + 255) / 256, 256, 0, stream>>>(hs0, Ahi, Alo, nA);
  splitT_kernel<<<(NC * DIM + 255) / 256, 256, 0, stream>>>(Wi1, Bhi, Blo);
  gemm_kernel<<<gemmBlocks, 256, 0, stream>>>(Ahi, Alo, Bhi, Blo, b1, Z);
  scan_kernel<<<NWG, SCAN_THREADS, 0, stream>>>(Z, Wh1, nullptr, hbufB, &ctr[1], xs, out, 1);
}
